// FasterRCNN_79929341378760
// MI455X (gfx1250) — matrix operand order repaired and checked by reading, not compiled, not run on hardware
//
#include <hip/hip_runtime.h>
#include <hip/hip_bf16.h>
#include <stdint.h>

// ---------------------------------------------------------------------------
// Faster R-CNN head for MI455X (gfx1250, wave32, WMMA).
// conv1 implicit-GEMM + fc1 + fc2 use v_wmma_f32_16x16x32_bf16 with
// fragment-order LDS tiles (b128 LDS traffic only).
// ---------------------------------------------------------------------------

typedef __attribute__((ext_vector_type(16))) __bf16          v16bf;
typedef __attribute__((ext_vector_type(8)))  unsigned int    v8u;
typedef __attribute__((ext_vector_type(8)))  float           v8f;

#define FHT     38
#define FWD     38
#define NPIX    1444           // 38*38
#define NPIXP   1472           // padded to 46*32
#define C_INCH  2048
#define C_MIDCH 512
#define NA      9
#define NANCH   12996          // NPIX * NA
#define PRE_NMS 2000
#define POST_NMS 300
#define KFC1    100352         // 2048*49
#define FCD     512
#define NSORT   16384

static __device__ __forceinline__ unsigned short f2bf(float f) {
  unsigned int u = __float_as_uint(f);
  u += 0x7FFFu + ((u >> 16) & 1u);     // round-to-nearest-even
  return (unsigned short)(u >> 16);
}
static __device__ __forceinline__ unsigned int pk2(float a, float b) {
  return (unsigned int)f2bf(a) | ((unsigned int)f2bf(b) << 16);
}

// ---------------------------------------------------------------------------
// K0: conv1_w (512,2048,3,3) f32 -> bf16 wT[kk][oc][ic]  (K-contiguous)
// ---------------------------------------------------------------------------
__global__ void wrepack_kernel(const float* __restrict__ w,
                               unsigned short* __restrict__ wT) {
  int i = blockIdx.x * 256 + threadIdx.x;
  if (i >= 9 * C_MIDCH * C_INCH) return;
  int kk  = i / (C_MIDCH * C_INCH);
  int rem = i % (C_MIDCH * C_INCH);
  int oc = rem / C_INCH, ic = rem % C_INCH;
  wT[i] = f2bf(w[((size_t)oc * C_INCH + ic) * 9 + kk]);
}

// ---------------------------------------------------------------------------
// K1: im2col with transpose: featsh[kk][p][ic] bf16, p padded to 1472 (zeros).
// 32x32 (p x ic) tiles through LDS so both global sides stay coalesced.
// ---------------------------------------------------------------------------
__global__ void im2col_kernel(const float* __restrict__ feat,
                              unsigned short* __restrict__ featsh) {
  __shared__ unsigned short t[32][33];
  const int icb = blockIdx.x * 32;
  const int pb  = blockIdx.y * 32;
  const int kk  = blockIdx.z;
  const int ky = kk / 3, kx = kk % 3;
  const int tid = threadIdx.x;
  for (int e = tid; e < 1024; e += 256) {
    int pl = e & 31, icl = e >> 5;
    int p = pb + pl;
    unsigned short h = 0;
    if (p < NPIX) {
      int py = p / FWD, px = p - py * FWD;
      int yy = py + ky - 1, xx = px + kx - 1;
      if (yy >= 0 && yy < FHT && xx >= 0 && xx < FWD)
        h = f2bf(feat[((size_t)(icb + icl) * FHT + yy) * FWD + xx]);
    }
    t[pl][icl] = h;
  }
  __syncthreads();
  for (int e = tid; e < 1024; e += 256) {
    int icl = e & 31, pl = e >> 5;
    featsh[((size_t)kk * NPIXP + pb + pl) * C_INCH + icb + icl] = t[pl][icl];
  }
}

// ---------------------------------------------------------------------------
// K2: conv1 3x3 + ReLU as implicit GEMM.  M=1472(px) x N=512(oc), K=9*2048.
// Block 256 thr = 8 waves; tile 32(M) x 128(N); 2 accumulators per wave.
// LDS tiles stored in WMMA fragment order -> b128 loads/stores only.
// ---------------------------------------------------------------------------
__global__ void __launch_bounds__(256) conv1_wmma_kernel(
    const unsigned short* __restrict__ featsh,  // [9][1472][2048] bf16
    const unsigned short* __restrict__ wT,      // [9][512][2048]  bf16
    const float* __restrict__ bias,
    float* __restrict__ mid2) {                 // [1444][512] f32
  __shared__ __align__(16) unsigned short lAf[2][32][16];  // [wm][lane][j]
  __shared__ __align__(16) unsigned short lBf[8][32][16];  // [ntile][lane][j]
  const int tid  = threadIdx.x;
  const int lane = tid & 31, wave = tid >> 5;
  const int wm = wave >> 2, wn = wave & 3;     // 2 M-tiles x 4 N-tile pairs
  const int nb = blockIdx.x * 128;             // oc base
  const int mb = blockIdx.y * 32;              // px base
  const int nB = tid >> 1, kgB = tid & 1;      // B chunk: 128 n x 2 kg
  v8f acc0 = {}, acc1 = {};
  for (int kq = 0; kq < 9 * C_INCH; kq += 32) {
    const int kk = kq >> 11;
    const int kc = kq & (C_INCH - 1);
    {  // B tile: wT rows, K-contiguous -> one 16-elem run per chunk
      const uint4* s = (const uint4*)(wT + ((size_t)kk * C_MIDCH + nb + nB) * C_INCH + kc + kgB * 16);
      uint4 lo = s[0], hi = s[1];
      unsigned short* d = &lBf[nB >> 4][(nB & 15) + 16 * kgB][0];
      *(uint4*)d = lo;
      *(uint4*)(d + 8) = hi;
    }
    if (tid < 64) {  // A tile: 32 px x 2 kg; K-contiguous -> one 16-elem run per chunk
      int m = tid >> 1, kg = tid & 1;
      const uint4* s = (const uint4*)(featsh + ((size_t)kk * NPIXP + mb + m) * C_INCH + kc + kg * 16);
      uint4 lo = s[0], hi = s[1];
      *(uint4*)&lAf[m >> 4][(m & 15) + 16 * kg][0] = lo;
      *(uint4*)&lAf[m >> 4][(m & 15) + 16 * kg][8] = hi;
    }
    __syncthreads();
    v8u au  = *(const v8u*)&lAf[wm][lane][0];
    v8u bu0 = *(const v8u*)&lBf[wn][lane][0];
    v8u bu1 = *(const v8u*)&lBf[wn + 4][lane][0];
    acc0 = __builtin_amdgcn_wmma_f32_16x16x32_bf16(
        false, __builtin_bit_cast(v16bf, au), false, __builtin_bit_cast(v16bf, bu0),
        (short)0, acc0, false, false);
    acc1 = __builtin_amdgcn_wmma_f32_16x16x32_bf16(
        false, __builtin_bit_cast(v16bf, au), false, __builtin_bit_cast(v16bf, bu1),
        (short)0, acc1, false, false);
    __syncthreads();
  }
  const int oc0 = nb + wn * 16 + (lane & 15);
  const int oc1 = oc0 + 64;
  const int pBase = mb + wm * 16 + ((lane & 16) ? 8 : 0);
  const float b0 = bias[oc0], b1 = bias[oc1];
#pragma unroll
  for (int v = 0; v < 8; ++v) {
    int p = pBase + v;
    if (p < NPIX) {
      mid2[(size_t)p * C_MIDCH + oc0] = fmaxf(acc0[v] + b0, 0.f);
      mid2[(size_t)p * C_MIDCH + oc1] = fmaxf(acc1[v] + b1, 0.f);
    }
  }
}

// ---------------------------------------------------------------------------
// K3: RPN 1x1 score + loc heads, softmax fg, anchor decode, min-size filter.
// mid2 is [p][512]; each thread reads its row with float4 loads (L2-resident).
// ---------------------------------------------------------------------------
__global__ void rpn_head_kernel(const float* __restrict__ mid2,
                                const float* __restrict__ scw, const float* __restrict__ scb,
                                const float* __restrict__ lcw, const float* __restrict__ lcb,
                                float* __restrict__ props, float* __restrict__ scores) {
  int idx = blockIdx.x * 256 + threadIdx.x;
  if (idx >= NA * NPIX) return;
  int a = idx / NPIX, p = idx % NPIX;
  float s0 = scb[2 * a], s1 = scb[2 * a + 1];
  float l0 = lcb[4 * a], l1 = lcb[4 * a + 1], l2 = lcb[4 * a + 2], l3 = lcb[4 * a + 3];
  const float4* mp = (const float4*)(mid2 + (size_t)p * C_MIDCH);
  const float4* w0 = (const float4*)(scw + (size_t)(2 * a) * C_MIDCH);
  const float4* w1 = (const float4*)(scw + (size_t)(2 * a + 1) * C_MIDCH);
  const float4* v0 = (const float4*)(lcw + (size_t)(4 * a) * C_MIDCH);
  const float4* v1 = (const float4*)(lcw + (size_t)(4 * a + 1) * C_MIDCH);
  const float4* v2 = (const float4*)(lcw + (size_t)(4 * a + 2) * C_MIDCH);
  const float4* v3 = (const float4*)(lcw + (size_t)(4 * a + 3) * C_MIDCH);
  for (int c = 0; c < C_MIDCH / 4; ++c) {
    float4 m = mp[c];
    float4 a0 = w0[c], a1 = w1[c], b0 = v0[c], b1 = v1[c], b2 = v2[c], b3 = v3[c];
    s0 += m.x * a0.x + m.y * a0.y + m.z * a0.z + m.w * a0.w;
    s1 += m.x * a1.x + m.y * a1.y + m.z * a1.z + m.w * a1.w;
    l0 += m.x * b0.x + m.y * b0.y + m.z * b0.z + m.w * b0.w;
    l1 += m.x * b1.x + m.y * b1.y + m.z * b1.z + m.w * b1.w;
    l2 += m.x * b2.x + m.y * b2.y + m.z * b2.z + m.w * b2.w;
    l3 += m.x * b3.x + m.y * b3.y + m.z * b3.z + m.w * b3.w;
  }
  float mx = fmaxf(s0, s1);
  float e0 = expf(s0 - mx), e1 = expf(s1 - mx);
  float fg = e1 / (e0 + e1);
  int py = p / FWD, px = p - py * FWD;
  int ri = a / 3, si = a % 3;
  float rat = (ri == 0) ? 0.5f : (ri == 1 ? 1.f : 2.f);
  float scl = (si == 0) ? 8.f : (si == 1 ? 16.f : 32.f);
  float hh = scl * sqrtf(rat);
  float wv = scl * sqrtf(1.f / rat);
  float sx = (float)(px * 16), sy = (float)(py * 16);
  float A0 = sx - hh * 0.5f, A1 = sy - wv * 0.5f;
  float A2 = sx + hh * 0.5f, A3 = sy + wv * 0.5f;
  float x = l0 * 16.f + A0;
  float y = l1 * 16.f + A1;
  float bw = expf(l2) * (A2 - A0);
  float bh = expf(l3) * (A3 - A1);
  float px1 = fminf(fmaxf(x, 0.f), 608.f);
  float px2 = fminf(fmaxf(x + bw, 0.f), 608.f);
  float py1 = fminf(fmaxf(y, 0.f), 608.f);
  float py2 = fminf(fmaxf(y + bh, 0.f), 608.f);
  bool ok = (px2 - px1 + 1.f >= 16.f) && (py2 - py1 + 1.f >= 16.f);
  int i = p * NA + a;
  props[i * 4 + 0] = px1; props[i * 4 + 1] = py1;
  props[i * 4 + 2] = px2; props[i * 4 + 3] = py2;
  scores[i] = ok ? fg : -1e30f;
}

// ---------------------------------------------------------------------------
// K4: top-2000 via single-workgroup bitonic sort (16384 keys, 128 KB dynamic
// LDS — legal on gfx1250's 320 KB/WGP).  Key = (~monotone(score))<<32 | idx.
// ---------------------------------------------------------------------------
__global__ void topk_sort_kernel(const float* __restrict__ scores,
                                 const float* __restrict__ props,
                                 int* __restrict__ top_idx,
                                 float* __restrict__ tb,
                                 int* __restrict__ tvalid) {
  extern __shared__ unsigned long long skey[];
  for (int i = threadIdx.x; i < NSORT; i += 1024) {
    unsigned long long key = 0xFFFFFFFFFFFFFFFFull;
    if (i < NANCH) {
      unsigned int b = __float_as_uint(scores[i]);
      unsigned int mono = (b & 0x80000000u) ? ~b : (b | 0x80000000u);
      unsigned int d = ~mono;                     // ascending sort -> descending score
      key = ((unsigned long long)d << 32) | (unsigned int)i;
    }
    skey[i] = key;
  }
  __syncthreads();
  for (int k = 2; k <= NSORT; k <<= 1) {
    for (int j = k >> 1; j > 0; j >>= 1) {
      for (int t = threadIdx.x; t < NSORT; t += 1024) {
        int ixj = t ^ j;
        if (ixj > t) {
          unsigned long long va = skey[t], vb = skey[ixj];
          bool asc = ((t & k) == 0);
          if (asc ? (va > vb) : (va < vb)) { skey[t] = vb; skey[ixj] = va; }
        }
      }
      __syncthreads();
    }
  }
  for (int i = threadIdx.x; i < PRE_NMS; i += 1024) {
    int id = (int)(skey[i] & 0xFFFFFFFFull);
    top_idx[i] = id;
    float s = scores[id];
    tvalid[i] = (s > -5e29f) ? 1 : 0;
    tb[i * 4 + 0] = props[id * 4 + 0];
    tb[i * 4 + 1] = props[id * 4 + 1];
    tb[i * 4 + 2] = props[id * 4 + 2];
    tb[i * 4 + 3] = props[id * 4 + 3];
  }
}

// ---------------------------------------------------------------------------
// K5: sequential-suppression NMS, one workgroup (32 waves), boxes in LDS.
// ---------------------------------------------------------------------------
__global__ void nms_kernel(const float* __restrict__ tb, const int* __restrict__ tvalid,
                           int* __restrict__ keep) {
  __shared__ float sX1[PRE_NMS], sY1[PRE_NMS], sX2[PRE_NMS], sY2[PRE_NMS], sAr[PRE_NMS];
  __shared__ unsigned char sSupp[PRE_NMS], sKeep[PRE_NMS];
  __shared__ int s_cur;
  const int tid = threadIdx.x;
  for (int i = tid; i < PRE_NMS; i += 1024) {
    float x1 = tb[i * 4 + 0], y1 = tb[i * 4 + 1];
    float x2 = tb[i * 4 + 2], y2 = tb[i * 4 + 3];
    sX1[i] = x1; sY1[i] = y1; sX2[i] = x2; sY2[i] = y2;
    sAr[i] = (x2 - x1 + 1.f) * (y2 - y1 + 1.f);
    sSupp[i] = 0; sKeep[i] = 0;
  }
  __syncthreads();
  for (int i = 0; i < PRE_NMS; ++i) {
    if (tid == 0) {
      int c = (!sSupp[i]) && tvalid[i];
      s_cur = c;
      sKeep[i] = (unsigned char)c;
    }
    __syncthreads();
    if (s_cur) {
      float xi1 = sX1[i], yi1 = sY1[i], xi2 = sX2[i], yi2 = sY2[i], ai = sAr[i];
      for (int jj = tid; jj < PRE_NMS; jj += 1024) {
        float xx1 = fmaxf(xi1, sX1[jj]);
        float yy1 = fmaxf(yi1, sY1[jj]);
        float xx2 = fminf(xi2, sX2[jj]);
        float yy2 = fminf(yi2, sY2[jj]);
        float inter = fmaxf(0.f, xx2 - xx1 + 1.f) * fmaxf(0.f, yy2 - yy1 + 1.f);
        float iou = inter / (ai + sAr[jj] - inter);
        if (iou > 0.5f) sSupp[jj] = 1;
      }
    }
    __syncthreads();
  }
  for (int i = tid; i < PRE_NMS; i += 1024) keep[i] = sKeep[i];
}

// ---------------------------------------------------------------------------
// K6: compact kept boxes -> rois (d_out), swapped rois + validity for align.
// ---------------------------------------------------------------------------
__global__ void select_kernel(const int* __restrict__ keep, const float* __restrict__ tb,
                              float* __restrict__ rois_sw, int* __restrict__ roi_valid,
                              float* __restrict__ out) {
  __shared__ int ssel[POST_NMS];
  __shared__ int snk;
  if (threadIdx.x == 0) {
    int pos = 0;
    for (int i = 0; i < PRE_NMS; ++i)
      if (keep[i]) { if (pos < POST_NMS) ssel[pos] = i; pos++; }
    snk = pos < POST_NMS ? pos : POST_NMS;
  }
  __syncthreads();
  for (int r = threadIdx.x; r < POST_NMS; r += blockDim.x) {
    int v = (r < snk) ? 1 : 0;
    roi_valid[r] = v;
    float b0 = 0.f, b1 = 0.f, b2 = 0.f, b3 = 0.f;
    if (v) {
      int i = ssel[r];
      b0 = tb[i * 4 + 0]; b1 = tb[i * 4 + 1];
      b2 = tb[i * 4 + 2]; b3 = tb[i * 4 + 3];
    }
    out[12000 + r * 4 + 0] = b0;
    out[12000 + r * 4 + 1] = b1;
    out[12000 + r * 4 + 2] = b2;
    out[12000 + r * 4 + 3] = b3;
    out[13200 + r] = 0.f;                        // roi_indices
    rois_sw[r * 4 + 0] = b1; rois_sw[r * 4 + 1] = b0;
    rois_sw[r * 4 + 2] = b3; rois_sw[r * 4 + 3] = b2;
  }
}

// ---------------------------------------------------------------------------
// K7: ROI-align max pool -> bf16 [300][2048*49] (feat lives in L2).
// ---------------------------------------------------------------------------
__global__ void roi_align_kernel(const float* __restrict__ feat,
                                 const float* __restrict__ rois_sw,
                                 const int* __restrict__ roi_valid,
                                 unsigned short* __restrict__ pool) {
  int idx = blockIdx.x * 256 + threadIdx.x;
  if (idx >= POST_NMS * KFC1) return;
  int r = idx / KFC1;
  int rem = idx % KFC1;
  int c = rem / 49, s = rem % 49;
  int ph = s / 7, pw = s % 7;
  float outv = 0.f;
  if (roi_valid[r]) {
    const float* roi = rois_sw + r * 4;
    int sw = (int)rintf(roi[0] * 0.0625f);
    int sh = (int)rintf(roi[1] * 0.0625f);
    int ew = (int)rintf(roi[2] * 0.0625f);
    int eh = (int)rintf(roi[3] * 0.0625f);
    float bw = (float)max(ew - sw, 1) / 7.f;
    float bh = (float)max(eh - sh, 1) / 7.f;
    int hs = min(max((int)floorf(ph * bh) + sh, 0), FHT);
    int he = min(max((int)ceilf((ph + 1) * bh) + sh, 0), FHT);
    int ws = min(max((int)floorf(pw * bw) + sw, 0), FWD);
    int we = min(max((int)ceilf((pw + 1) * bw) + sw, 0), FWD);
    if (he > hs && we > ws) {
      float m = -1e30f;
      const float* fc = feat + (size_t)c * NPIX;
      for (int y = hs; y < he; ++y)
        for (int x = ws; x < we; ++x)
          m = fmaxf(m, fc[y * FWD + x]);
      outv = m;
    }
  }
  pool[idx] = f2bf(outv);
}

// ---------------------------------------------------------------------------
// K8/K9: WMMA GEMM  D[r][n] = act( sum_k A[r][k]*B[n][k] + bias[n] ).
// A bf16 [Mvalid][K]; B f32 row-major [N][K] (converted on the fly).
// Block 256 thr = 8 waves; tile 16(M) x 256(N); 2 accumulators per wave.
// ---------------------------------------------------------------------------
__global__ void __launch_bounds__(256) gemm_wmma_kernel(
    const unsigned short* __restrict__ A, const float* __restrict__ B,
    const float* __restrict__ bias,
    unsigned short* __restrict__ outBf, float* __restrict__ outF,
    int Mvalid, int K, int N, int do_relu) {
  __shared__ __align__(16) unsigned short lAf[32][16];       // [lane][j]
  __shared__ __align__(16) unsigned short lBf[16][32][16];   // [ntile][lane][j]
  const int tid  = threadIdx.x;
  const int lane = tid & 31, wave = tid >> 5;
  const int nb = blockIdx.x * 256;
  const int rb = blockIdx.y * 16;
  v8f acc0 = {}, acc1 = {};
  for (int kc = 0; kc < K; kc += 32) {
#pragma unroll
    for (int h = 0; h < 2; ++h) {                 // 512 B chunks, 2 per thread
      int chunk = tid + h * 256;
      int n = chunk >> 1, kg = chunk & 1;
      const float4* s = (const float4*)(B + (size_t)(nb + n) * K + kc + kg * 16);
      float4 f0 = s[0], f1 = s[1], f2v = s[2], f3 = s[3];
      if (kc + 32 < K) __builtin_prefetch((const void*)(s + 8), 0, 1);
      uint4 lo, hi;
      lo.x = pk2(f0.x, f0.y);  lo.y = pk2(f0.z, f0.w);
      lo.z = pk2(f1.x, f1.y);  lo.w = pk2(f1.z, f1.w);
      hi.x = pk2(f2v.x, f2v.y); hi.y = pk2(f2v.z, f2v.w);
      hi.z = pk2(f3.x, f3.y);  hi.w = pk2(f3.z, f3.w);
      unsigned short* d = &lBf[n >> 4][(n & 15) + 16 * kg][0];
      *(uint4*)d = lo;
      *(uint4*)(d + 8) = hi;
    }
    if (tid < 32) {                               // 32 A chunks (16 m x 2 kg)
      int m = tid >> 1, kg = tid & 1;
      int r = rb + m;
      uint4 lo = make_uint4(0u, 0u, 0u, 0u), hi = lo;
      if (r < Mvalid) {
        const uint4* s = (const uint4*)(A + (size_t)r * K + kc + kg * 16);
        lo = s[0]; hi = s[1];
      }
      *(uint4*)&lAf[m + 16 * kg][0] = lo;
      *(uint4*)&lAf[m + 16 * kg][8] = hi;
    }
    __syncthreads();
    v8u au  = *(const v8u*)&lAf[lane][0];
    v8u bu0 = *(const v8u*)&lBf[wave][lane][0];
    v8u bu1 = *(const v8u*)&lBf[wave + 8][lane][0];
    acc0 = __builtin_amdgcn_wmma_f32_16x16x32_bf16(
        false, __builtin_bit_cast(v16bf, au), false, __builtin_bit_cast(v16bf, bu0),
        (short)0, acc0, false, false);
    acc1 = __builtin_amdgcn_wmma_f32_16x16x32_bf16(
        false, __builtin_bit_cast(v16bf, au), false, __builtin_bit_cast(v16bf, bu1),
        (short)0, acc1, false, false);
    __syncthreads();
  }
  const int n0 = nb + wave * 16 + (lane & 15);
  const int n1 = n0 + 128;
  const int rbase = rb + ((lane & 16) ? 8 : 0);
  const float b0 = bias[n0], b1 = bias[n1];
#pragma unroll
  for (int v = 0; v < 8; ++v) {
    int r = rbase + v;
    if (r < Mvalid) {
      float x0 = acc0[v] + b0, x1 = acc1[v] + b1;
      if (do_relu) { x0 = fmaxf(x0, 0.f); x1 = fmaxf(x1, 0.f); }
      if (outBf) {
        outBf[(size_t)r * N + n0] = f2bf(x0);
        outBf[(size_t)r * N + n1] = f2bf(x1);
      }
      if (outF) {
        outF[(size_t)r * N + n0] = x0;
        outF[(size_t)r * N + n1] = x1;
      }
    }
  }
}

// ---------------------------------------------------------------------------
// K10: tiny classification/regression heads (300 x 40 dots of length 512).
// ---------------------------------------------------------------------------
__global__ void heads_kernel(const float* __restrict__ f2,
                             const float* __restrict__ clw, const float* __restrict__ clb,
                             const float* __restrict__ csw, const float* __restrict__ csb,
                             float* __restrict__ out) {
  int idx = blockIdx.x * 256 + threadIdx.x;
  if (idx >= POST_NMS * 40) return;
  int r = idx / 40, o = idx % 40;
  const float4* f = (const float4*)(f2 + (size_t)r * FCD);
  const float4* w;
  float d;
  if (o < 32) { w = (const float4*)(clw + (size_t)o * FCD); d = clb[o]; }
  else        { w = (const float4*)(csw + (size_t)(o - 32) * FCD); d = csb[o - 32]; }
  for (int c = 0; c < FCD / 4; ++c) {
    float4 a = f[c], b = w[c];
    d += a.x * b.x + a.y * b.y + a.z * b.z + a.w * b.w;
  }
  if (o < 32) out[r * 32 + o] = d;
  else        out[9600 + r * 8 + (o - 32)] = d;
}

// ---------------------------------------------------------------------------
extern "C" void kernel_launch(void* const* d_in, const int* in_sizes, int n_in,
                              void* d_out, int out_size, void* d_ws, size_t ws_size,
                              hipStream_t stream) {
  (void)in_sizes; (void)n_in; (void)out_size; (void)ws_size;
  const float* feat        = (const float*)d_in[0];
  const float* conv1_w     = (const float*)d_in[1];
  const float* conv1_b     = (const float*)d_in[2];
  const float* score_w     = (const float*)d_in[3];
  const float* score_b     = (const float*)d_in[4];
  const float* loc_w       = (const float*)d_in[5];
  const float* loc_b       = (const float*)d_in[6];
  const float* fc1_w       = (const float*)d_in[7];
  const float* fc1_b       = (const float*)d_in[8];
  const float* fc2_w       = (const float*)d_in[9];
  const float* fc2_b       = (const float*)d_in[10];
  const float* cls_loc_w   = (const float*)d_in[11];
  const float* cls_loc_b   = (const float*)d_in[12];
  const float* cls_score_w = (const float*)d_in[13];
  const float* cls_score_b = (const float*)d_in[14];
  float* out = (float*)d_out;

  char* ws = (char*)d_ws;
  size_t off = 0;
  auto take = [&](size_t bytes) -> char* {
    char* p = ws + off;
    off += (bytes + 255) & ~(size_t)255;
    return p;
  };
  unsigned short* featsh = (unsigned short*)take((size_t)9 * NPIXP * C_INCH * 2);
  unsigned short* wT     = (unsigned short*)take((size_t)9 * C_MIDCH * C_INCH * 2);
  float* mid2            = (float*)take((size_t)NPIX * C_MIDCH * 4);
  float* props           = (float*)take((size_t)NANCH * 4 * 4);
  float* scores          = (float*)take((size_t)NANCH * 4);
  int*   top_idx         = (int*)take(PRE_NMS * 4);
  float* tb              = (float*)take(PRE_NMS * 4 * 4);
  int*   tvalid          = (int*)take(PRE_NMS * 4);
  int*   keep            = (int*)take(PRE_NMS * 4);
  float* rois_sw         = (float*)take(POST_NMS * 4 * 4);
  int*   roi_valid       = (int*)take(POST_NMS * 4);
  unsigned short* pool   = (unsigned short*)take((size_t)POST_NMS * KFC1 * 2);
  unsigned short* f1     = (unsigned short*)take((size_t)POST_NMS * FCD * 2);
  float* f2              = (float*)take((size_t)POST_NMS * FCD * 4);

  wrepack_kernel<<<(9 * C_MIDCH * C_INCH + 255) / 256, 256, 0, stream>>>(conv1_w, wT);
  im2col_kernel<<<dim3(64, 46, 9), 256, 0, stream>>>(feat, featsh);
  conv1_wmma_kernel<<<dim3(4, 46), 256, 0, stream>>>(featsh, wT, conv1_b, mid2);
  rpn_head_kernel<<<(NANCH + 255) / 256, 256, 0, stream>>>(
      mid2, score_w, score_b, loc_w, loc_b, props, scores);
  topk_sort_kernel<<<1, 1024, NSORT * sizeof(unsigned long long), stream>>>(
      scores, props, top_idx, tb, tvalid);
  nms_kernel<<<1, 1024, 0, stream>>>(tb, tvalid, keep);
  select_kernel<<<1, 512, 0, stream>>>(keep, tb, rois_sw, roi_valid, out);
  roi_align_kernel<<<(POST_NMS * KFC1 + 255) / 256, 256, 0, stream>>>(
      feat, rois_sw, roi_valid, pool);
  gemm_wmma_kernel<<<dim3(2, 19), 256, 0, stream>>>(
      pool, fc1_w, fc1_b, f1, nullptr, POST_NMS, KFC1, FCD, 1);
  gemm_wmma_kernel<<<dim3(2, 19), 256, 0, stream>>>(
      f1, fc2_w, fc2_b, nullptr, f2, POST_NMS, FCD, FCD, 1);
  heads_kernel<<<(POST_NMS * 40 + 255) / 256, 256, 0, stream>>>(
      f2, cls_loc_w, cls_loc_b, cls_score_w, cls_score_b, out);
}
